// SpatialTemporalEnhancement_65154653880837
// MI455X (gfx1250) — matrix operand order repaired and checked by reading, not compiled, not run on hardware
//
#include <hip/hip_runtime.h>
#include <hip/hip_bf16.h>

typedef __attribute__((ext_vector_type(16))) _Float16 v16h;
typedef __attribute__((ext_vector_type(8)))  _Float16 v8h;
typedef __attribute__((ext_vector_type(8)))  float    v8f;
typedef __attribute__((__vector_size__(4 * sizeof(int)))) int vi4;

#define BB     4
#define TT     16
#define DMODEL 256
#define HH     32
#define WWID   32
#define SEQS   (BB * HH * WWID)        // 4096
#define MROWS  (SEQS * TT)             // 65536
#define DINNER 512
#define DSTATE 16
#define DTRANK 16
#define NHEADS 8
#define HDIM   32

// ---------------------------------------------------------------------------
// CDNA5 async global->LDS copy (GLOBAL_LOAD_ASYNC_TO_LDS_B128, ASYNCcnt).
// Falls back to a synchronous VGPR round-trip if the builtin is absent.
// ---------------------------------------------------------------------------
__device__ __forceinline__ void async_cp16B(const _Float16* g, _Float16* s) {
#if __has_builtin(__builtin_amdgcn_global_load_async_to_lds_b128)
  __builtin_amdgcn_global_load_async_to_lds_b128((vi4*)g, (vi4*)s, 0, 0);
#else
  *(uint4*)s = *(const uint4*)g;
#endif
}
__device__ __forceinline__ void wait_async0() {
#if __has_builtin(__builtin_amdgcn_global_load_async_to_lds_b128)
#if __has_builtin(__builtin_amdgcn_s_wait_asynccnt)
  __builtin_amdgcn_s_wait_asynccnt(0);
#else
  asm volatile("s_wait_asynccnt 0x0" ::: "memory");
#endif
#endif
}

// ---------------------------------------------------------------------------
// Fragment loaders (CDNA5 WMMA f16 layouts, wave32)
// A (16x32): half j -> K = 16*(j>>3) + 8*(lane>>4) + (j&7), M = lane&15
// B (32x16): half j -> K = 16*(j>>3) + 8*(lane>>4) + (j&7), N = lane&15
// ---------------------------------------------------------------------------
__device__ __forceinline__ v16h cat8(v8h lo, v8h hi) {
  return __builtin_shufflevector(lo, hi, 0,1,2,3,4,5,6,7,8,9,10,11,12,13,14,15);
}
__device__ __forceinline__ v16h fragA_lds(const _Float16* rowp, int lane) {
  int ko = (lane >> 4) * 8;
  v8h lo = *(const v8h*)(rowp + ko);
  v8h hi = *(const v8h*)(rowp + 16 + ko);
  return cat8(lo, hi);
}
__device__ __forceinline__ v16h fragB_lds(const _Float16* rowp, int lane) {
  int ko = (lane >> 4) * 8;
  v8h lo = *(const v8h*)(rowp + ko);
  v8h hi = *(const v8h*)(rowp + 16 + ko);
  return cat8(lo, hi);
}

// ---------------------------------------------------------------------------
// Pack E (B,T,D,H,W) -> row-major (MROWS x DMODEL), f32 + f16
// ---------------------------------------------------------------------------
__global__ void k_packE(const float* __restrict__ E, float* __restrict__ Ef,
                        _Float16* __restrict__ Eh) {
  size_t i = (size_t)blockIdx.x * blockDim.x + threadIdx.x;
  if (i >= (size_t)MROWS * DMODEL) return;
  int d   = (int)(i & (DMODEL - 1));
  size_t row = i >> 8;
  int t   = (int)(row & (TT - 1));
  int seq = (int)(row >> 4);
  int w   = seq & (WWID - 1);
  int h   = (seq >> 5) & (HH - 1);
  int b   = seq >> 10;
  size_t src = ((((size_t)b * TT + t) * DMODEL + d) * HH + h) * WWID + w;
  float v = E[src];
  Ef[i] = v;
  Eh[i] = (_Float16)v;
}

__global__ void k_cvt(const float* __restrict__ s, _Float16* __restrict__ d, int n) {
  int i = blockIdx.x * blockDim.x + threadIdx.x;
  if (i < n) d[i] = (_Float16)s[i];
}

__global__ void k_cvt_pad(const float* __restrict__ s, _Float16* __restrict__ d,
                          int rows, int cols, int prows) {
  int i = blockIdx.x * blockDim.x + threadIdx.x;
  if (i >= prows * cols) return;
  int r = i / cols, c = i - r * cols;
  d[i] = (r < rows) ? (_Float16)s[(size_t)r * cols + c] : (_Float16)0.0f;
}

// ---------------------------------------------------------------------------
// Tiled WMMA GEMM: C(MxN,f32) [, Ch f16] = A(MxK,f16) * W(NxK,f16)^T + bias
// Block = 128 threads (4 waves); tile 64x64, k-step 32.
// Double-buffered LDS fed by async global->LDS DMA.
// ---------------------------------------------------------------------------
#define GBM 64
#define GBN 64
#define GBK 32

__global__ __launch_bounds__(128) void k_gemm(
    const _Float16* __restrict__ A, const _Float16* __restrict__ Wm,
    float* __restrict__ C, _Float16* __restrict__ Ch,
    const float* __restrict__ bias, int K, int ldc) {
  __shared__ alignas(16) _Float16 As[2][GBM * GBK];
  __shared__ alignas(16) _Float16 Ws[2][GBN * GBK];
  int tid  = threadIdx.x;
  int lane = tid & 31;
  int wave = tid >> 5;
  int bm = blockIdx.y * GBM;
  int bn = blockIdx.x * GBN;

  v8f acc[4] = {};
  int sr = tid >> 1;              // staged row 0..63
  int sc = (tid & 1) * 16;        // staged half-offset 0/16 (covers 16 halves)

  const _Float16* gA = A  + (size_t)(bm + sr) * K + sc;
  const _Float16* gW = Wm + (size_t)(bn + sr) * K + sc;

  // Preload k-tile 0 into buffer 0 (async DMA to LDS)
  {
    _Float16* dA = &As[0][sr * GBK + sc];
    _Float16* dW = &Ws[0][sr * GBK + sc];
    async_cp16B(gA,     dA);
    async_cp16B(gA + 8, dA + 8);
    async_cp16B(gW,     dW);
    async_cp16B(gW + 8, dW + 8);
  }
  wait_async0();
  __syncthreads();

  for (int k0 = 0; k0 < K; k0 += GBK) {
    int cur = (k0 >> 5) & 1;
    int nxt = cur ^ 1;
    if (k0 + GBK < K) {  // overlap: DMA next k-tile while WMMAs run on cur
      _Float16* dA = &As[nxt][sr * GBK + sc];
      _Float16* dW = &Ws[nxt][sr * GBK + sc];
      async_cp16B(gA + k0 + GBK,     dA);
      async_cp16B(gA + k0 + GBK + 8, dA + 8);
      async_cp16B(gW + k0 + GBK,     dW);
      async_cp16B(gW + k0 + GBK + 8, dW + 8);
    }
    if (k0 + 2 * GBK < K) {  // warm L2 two tiles ahead (global_prefetch_b8)
      __builtin_prefetch(gA + k0 + 2 * GBK, 0, 3);
      __builtin_prefetch(gW + k0 + 2 * GBK, 0, 3);
    }
    int m = wave * 16 + (lane & 15);
    v16h af = fragA_lds(&As[cur][m * GBK], lane);
#pragma unroll
    for (int nt = 0; nt < 4; ++nt) {
      int n = nt * 16 + (lane & 15);
      v16h wf = fragB_lds(&Ws[cur][n * GBK], lane);
      acc[nt] = __builtin_amdgcn_wmma_f32_16x16x32_f16(
          false, af, false, wf, (short)0, acc[nt], false, false);
    }
    wait_async0();     // this wave's DMAs landed
    __syncthreads();   // all waves' DMAs landed; cur free for reuse
  }

  int col0  = bn + (lane & 15);
  int rbase = bm + wave * 16 + ((lane >> 4) * 8);
#pragma unroll
  for (int nt = 0; nt < 4; ++nt) {
    int col = col0 + nt * 16;
    float bi = bias ? bias[col] : 0.0f;
#pragma unroll
    for (int r = 0; r < 8; ++r) {
      float v = acc[nt][r] + bi;
      size_t idx = (size_t)(rbase + r) * ldc + col;
      if (C)  C[idx]  = v;
      if (Ch) Ch[idx] = (_Float16)v;
    }
  }
}

// ---------------------------------------------------------------------------
// Causal depthwise conv (k=4) + SiLU ; reads xm half of xz (ld 1024)
// ---------------------------------------------------------------------------
__global__ void k_conv(const float* __restrict__ xz, const float* __restrict__ cw,
                       const float* __restrict__ cb, float* __restrict__ xc,
                       _Float16* __restrict__ xch) {
  size_t i = (size_t)blockIdx.x * blockDim.x + threadIdx.x;
  if (i >= (size_t)MROWS * DINNER) return;
  int ch = (int)(i & (DINNER - 1));
  size_t row = i >> 9;
  int t = (int)(row & (TT - 1));
  size_t base = (row - t) * 1024 + ch;
  float acc = cb[ch];
#pragma unroll
  for (int j = 0; j < 4; ++j) {
    int tt = t - 3 + j;
    if (tt >= 0) acc += cw[ch * 4 + j] * xz[base + (size_t)tt * 1024];
  }
  float s = acc / (1.0f + __expf(-acc));
  xc[i]  = s;
  xch[i] = (_Float16)s;
}

// ---------------------------------------------------------------------------
// Selective scan: 1 block / sequence, 128 threads, 4 channels each.
// ---------------------------------------------------------------------------
__global__ __launch_bounds__(128) void k_scan(
    const float* __restrict__ dbc,   // MROWS x 64 (dt|B|C|pad)
    const float* __restrict__ xc,    // MROWS x 512
    const float* __restrict__ xz,    // MROWS x 1024 (z at +512)
    const float* __restrict__ dtw,   // 512 x 16
    const float* __restrict__ dtb, const float* __restrict__ Alog,
    const float* __restrict__ Dp, _Float16* __restrict__ g16) {
  __shared__ float Ash[DINNER * DSTATE];
  __shared__ float sh_dt[16], sh_B[16], sh_C[16];
  int tid = threadIdx.x;
  int seq = blockIdx.x;
  for (int i = tid; i < DINNER * DSTATE; i += 128) Ash[i] = -__expf(Alog[i]);
  int ch0 = tid * 4;
  float h[4][DSTATE];
  float dw[4][DTRANK], db4[4], Dp4[4];
#pragma unroll
  for (int c = 0; c < 4; ++c) {
    db4[c] = dtb[ch0 + c]; Dp4[c] = Dp[ch0 + c];
#pragma unroll
    for (int s = 0; s < DSTATE; ++s) h[c][s] = 0.0f;
    for (int r = 0; r < DTRANK; ++r) dw[c][r] = dtw[(ch0 + c) * DTRANK + r];
  }
  __syncthreads();
  for (int t = 0; t < TT; ++t) {
    size_t row = (size_t)seq * TT + t;
    if (tid < 48) {
      float v = dbc[row * 64 + tid];
      if (tid < 16)      sh_dt[tid] = v;
      else if (tid < 32) sh_B[tid - 16] = v;
      else               sh_C[tid - 32] = v;
    }
    __syncthreads();
#pragma unroll
    for (int c = 0; c < 4; ++c) {
      int ch = ch0 + c;
      float d = db4[c];
#pragma unroll
      for (int r = 0; r < DTRANK; ++r) d += sh_dt[r] * dw[c][r];
      d = (d > 20.0f) ? d : __logf(1.0f + __expf(d));   // softplus
      float x  = xc[row * DINNER + ch];
      float dx = d * x;
      float y = 0.0f;
      const float* Arow = Ash + ch * DSTATE;
#pragma unroll
      for (int s = 0; s < DSTATE; ++s) {
        float dA = __expf(d * Arow[s]);
        h[c][s] = dA * h[c][s] + dx * sh_B[s];
        y += h[c][s] * sh_C[s];
      }
      float z  = xz[row * 1024 + DINNER + ch];
      float sz = z / (1.0f + __expf(-z));
      g16[row * DINNER + ch] = (_Float16)((y + x * Dp4[c]) * sz);
    }
    __syncthreads();
  }
}

// ---------------------------------------------------------------------------
// Cross-attention: block = 8 waves = 8 heads of one sequence. WMMA scores,
// LDS softmax, WMMA attn*V.
// ---------------------------------------------------------------------------
__global__ __launch_bounds__(256) void k_attn(
    const _Float16* __restrict__ q16, const _Float16* __restrict__ k16,
    const _Float16* __restrict__ v16, _Float16* __restrict__ ctx) {
  __shared__ float sc[NHEADS][16 * 17];
  int tid  = threadIdx.x;
  int lane = tid & 31;
  int head = tid >> 5;
  size_t rb = (size_t)blockIdx.x * TT;
  int colb = head * HDIM;
  int m   = lane & 15;
  int hi16 = lane >> 4;

  const _Float16* qp = q16 + (rb + m) * DMODEL + colb;
  v16h qf = cat8(*(const v8h*)(qp + hi16 * 8), *(const v8h*)(qp + 16 + hi16 * 8));
  const _Float16* kp = k16 + (rb + m) * DMODEL + colb + hi16 * 8;
  v16h kf = cat8(*(const v8h*)(kp), *(const v8h*)(kp + 16));

  v8f s = {};
  s = __builtin_amdgcn_wmma_f32_16x16x32_f16(false, qf, false, kf, (short)0, s,
                                             false, false);
  const float scale = 0.17677669529663687f;  // 1/sqrt(32)
  float* my = &sc[head][0];
  int n = lane & 15;
#pragma unroll
  for (int r = 0; r < 8; ++r) my[(hi16 * 8 + r) * 17 + n] = s[r] * scale;
  __syncthreads();
  if (lane < 16) {  // softmax: lane handles one query row
    float* rowp = my + lane * 17;
    float mx = rowp[0];
    for (int j = 1; j < 16; ++j) mx = fmaxf(mx, rowp[j]);
    float sum = 0.0f;
    for (int j = 0; j < 16; ++j) { float e = __expf(rowp[j] - mx); rowp[j] = e; sum += e; }
    float inv = 1.0f / sum;
    for (int j = 0; j < 16; ++j) rowp[j] *= inv;
  }
  __syncthreads();
  v16h af = {};
  const float* ar = my + m * 17 + hi16 * 8;
#pragma unroll
  for (int j = 0; j < 8; ++j) af[j] = (_Float16)ar[j];
#pragma unroll
  for (int nt = 0; nt < 2; ++nt) {
    v16h vf;
#pragma unroll
    for (int j = 0; j < 16; ++j) {
      int k  = (j < 8) ? hi16 * 8 + j : 8 + hi16 * 8 + j;
      int kk = k < 16 ? k : 15;              // clamped; A is zero there
      vf[j] = v16[(rb + kk) * DMODEL + colb + nt * 16 + n];
    }
    v8f o = {};
    o = __builtin_amdgcn_wmma_f32_16x16x32_f16(false, af, false, vf, (short)0, o,
                                               false, false);
#pragma unroll
    for (int r = 0; r < 8; ++r)
      ctx[(rb + hi16 * 8 + r) * DMODEL + colb + nt * 16 + n] = (_Float16)o[r];
  }
}

// ---------------------------------------------------------------------------
// Residual + LayerNorm + transpose back to (B,T,D,H,W)
// ---------------------------------------------------------------------------
__global__ __launch_bounds__(256) void k_ln(
    const float* __restrict__ ao, const float* __restrict__ Ef,
    const float* __restrict__ g, const float* __restrict__ b,
    float* __restrict__ out) {
  __shared__ float red[8];
  __shared__ float stat;
  int d = threadIdx.x;
  size_t row = blockIdx.x;
  float x = ao[row * DMODEL + d] + Ef[row * DMODEL + d];
  float part = x;
  for (int o = 16; o; o >>= 1) part += __shfl_down(part, o, 32);
  if ((d & 31) == 0) red[d >> 5] = part;
  __syncthreads();
  if (d == 0) { float s = 0; for (int i = 0; i < 8; ++i) s += red[i]; stat = s / DMODEL; }
  __syncthreads();
  float mu = stat, dx = x - mu;
  __syncthreads();
  part = dx * dx;
  for (int o = 16; o; o >>= 1) part += __shfl_down(part, o, 32);
  if ((d & 31) == 0) red[d >> 5] = part;
  __syncthreads();
  if (d == 0) { float s = 0; for (int i = 0; i < 8; ++i) s += red[i]; stat = s / DMODEL; }
  __syncthreads();
  float r = rsqrtf(stat + 1e-5f);
  float y = dx * r * g[d] + b[d];
  int t = (int)(row & (TT - 1));
  int seq = (int)(row >> 4);
  int w = seq & (WWID - 1), h = (seq >> 5) & (HH - 1), bb = seq >> 10;
  out[((((size_t)bb * TT + t) * DMODEL + d) * HH + h) * WWID + w] = y;
}

// ---------------------------------------------------------------------------
extern "C" void kernel_launch(void* const* d_in, const int* in_sizes, int n_in,
                              void* d_out, int out_size, void* d_ws, size_t ws_size,
                              hipStream_t stream) {
  const float* E     = (const float*)d_in[0];
  const float* inw   = (const float*)d_in[1];
  const float* convw = (const float*)d_in[2];
  const float* convb = (const float*)d_in[3];
  const float* xprw  = (const float*)d_in[4];
  const float* dtw   = (const float*)d_in[5];
  const float* dtb   = (const float*)d_in[6];
  const float* Alog  = (const float*)d_in[7];
  const float* Dp    = (const float*)d_in[8];
  const float* moutw = (const float*)d_in[9];
  const float* qw = (const float*)d_in[10]; const float* qb = (const float*)d_in[11];
  const float* kw = (const float*)d_in[12]; const float* kb = (const float*)d_in[13];
  const float* vw = (const float*)d_in[14]; const float* vb = (const float*)d_in[15];
  const float* ow = (const float*)d_in[16]; const float* ob = (const float*)d_in[17];
  const float* lng = (const float*)d_in[18]; const float* lnb = (const float*)d_in[19];
  float* out = (float*)d_out;

  char* ws = (char*)d_ws;
  size_t off = 0;
  auto carve = [&](size_t bytes) -> void* {
    void* p = ws + off;
    off += (bytes + 255) & ~(size_t)255;
    return p;
  };
  float*    Ef32 = (float*)carve((size_t)MROWS * DMODEL * 4);
  _Float16* Eh   = (_Float16*)carve((size_t)MROWS * DMODEL * 2);
  _Float16* w_in = (_Float16*)carve((size_t)1024 * 256 * 2);
  _Float16* w_xp = (_Float16*)carve((size_t)64 * 512 * 2);
  _Float16* w_mo = (_Float16*)carve((size_t)256 * 512 * 2);
  _Float16* w_q  = (_Float16*)carve((size_t)256 * 256 * 2);
  _Float16* w_k  = (_Float16*)carve((size_t)256 * 256 * 2);
  _Float16* w_v  = (_Float16*)carve((size_t)256 * 256 * 2);
  _Float16* w_o  = (_Float16*)carve((size_t)256 * 256 * 2);
  float*    xz32 = (float*)carve((size_t)MROWS * 1024 * 4);
  float*    xc32 = (float*)carve((size_t)MROWS * DINNER * 4);
  _Float16* xc16 = (_Float16*)carve((size_t)MROWS * DINNER * 2);
  float*    dbc  = (float*)carve((size_t)MROWS * 64 * 4);
  _Float16* g16  = (_Float16*)carve((size_t)MROWS * DINNER * 2);
  _Float16* m16  = (_Float16*)carve((size_t)MROWS * DMODEL * 2);
  _Float16* q16  = (_Float16*)carve((size_t)MROWS * DMODEL * 2);
  _Float16* k16  = (_Float16*)carve((size_t)MROWS * DMODEL * 2);
  _Float16* v16p = (_Float16*)carve((size_t)MROWS * DMODEL * 2);
  _Float16* ctx  = (_Float16*)carve((size_t)MROWS * DMODEL * 2);
  float*    ao32 = (float*)carve((size_t)MROWS * DMODEL * 4);
  (void)ws_size; (void)in_sizes; (void)n_in; (void)out_size;

  k_packE<<<(MROWS * DMODEL) / 256, 256, 0, stream>>>(E, Ef32, Eh);
  k_cvt<<<(1024 * 256) / 256, 256, 0, stream>>>(inw, w_in, 1024 * 256);
  k_cvt<<<(256 * 512) / 256, 256, 0, stream>>>(moutw, w_mo, 256 * 512);
  k_cvt<<<(256 * 256) / 256, 256, 0, stream>>>(qw, w_q, 256 * 256);
  k_cvt<<<(256 * 256) / 256, 256, 0, stream>>>(kw, w_k, 256 * 256);
  k_cvt<<<(256 * 256) / 256, 256, 0, stream>>>(vw, w_v, 256 * 256);
  k_cvt<<<(256 * 256) / 256, 256, 0, stream>>>(ow, w_o, 256 * 256);
  k_cvt_pad<<<(64 * 512) / 256, 256, 0, stream>>>(xprw, w_xp, 48, 512, 64);
  // in_proj: xz = E @ in_w^T   (65536 x 1024, K=256)
  k_gemm<<<dim3(1024 / GBN, MROWS / GBM), 128, 0, stream>>>(
      Eh, w_in, xz32, nullptr, nullptr, 256, 1024);
  k_conv<<<((size_t)MROWS * DINNER) / 256, 256, 0, stream>>>(xz32, convw, convb,
                                                             xc32, xc16);
  // x_proj: dbc = xc @ x_proj^T  (N padded to 64, K=512)
  k_gemm<<<dim3(1, MROWS / GBM), 128, 0, stream>>>(xc16, w_xp, dbc, nullptr,
                                                   nullptr, 512, 64);
  k_scan<<<SEQS, 128, 0, stream>>>(dbc, xc32, xz32, dtw, dtb, Alog, Dp, g16);
  // out_proj: m = g @ mamba_out^T
  k_gemm<<<dim3(256 / GBN, MROWS / GBM), 128, 0, stream>>>(
      g16, w_mo, nullptr, m16, nullptr, 512, 256);
  // Q/K/V projections
  k_gemm<<<dim3(256 / GBN, MROWS / GBM), 128, 0, stream>>>(
      Eh, w_q, nullptr, q16, qb, 256, 256);
  k_gemm<<<dim3(256 / GBN, MROWS / GBM), 128, 0, stream>>>(
      m16, w_k, nullptr, k16, kb, 256, 256);
  k_gemm<<<dim3(256 / GBN, MROWS / GBM), 128, 0, stream>>>(
      m16, w_v, nullptr, v16p, vb, 256, 256);
  k_attn<<<SEQS, 256, 0, stream>>>(q16, k16, v16p, ctx);
  // output projection
  k_gemm<<<dim3(256 / GBN, MROWS / GBM), 128, 0, stream>>>(
      ctx, w_o, ao32, nullptr, ob, 256, 256);
  k_ln<<<MROWS, 256, 0, stream>>>(ao32, Ef32, lng, lnb, out);
}
